// InterDiffCov_89103391523711
// MI455X (gfx1250) — matrix operand order repaired and checked by reading, not compiled, not run on hardware
//
#include <hip/hip_runtime.h>

// ---------------------------------------------------------------------------
// Problem constants (from the reference): M=16, Vn=8192, DIN=DOUT=512.
// Plan:
//   k_prep_w : W[do][dk] f32 -> Wt[dk][do] bf16 (3 weights) into d_ws
//   k_qk_gram: fused Q/K GEMM (bf16 WMMA, f32 acc) + Gram G[16][16] and
//              sum(Q^2)[16] reduced to per-block partials (deterministic)
//   k_proj   : sigmoid(G/||Q_i||) -> 0.5*(softmax0+softmax1) -> 10x projection
//   k_v_out  : fused V GEMM (bf16 WMMA) + out[j] = sum_i P[i][j] V[i]
// ---------------------------------------------------------------------------

typedef __attribute__((ext_vector_type(16))) __bf16 v16bf;
typedef __attribute__((ext_vector_type(8)))  float  v8f;
typedef __attribute__((ext_vector_type(2)))  float  v2f;

#define MM   16
#define VN   8192
#define DD   512
#define NB1  256          // blocks for k_qk_gram / k_v_out
#define VC   32           // v's per block (NB1 * VC == VN)
#define NV   2            // v's per LDS group (weight-tile reuse factor)
#define NGRP (VC / NV)

static __device__ __forceinline__ unsigned pkbf(float a, float b) {
  union { __bf16 h[2]; unsigned u; } z;
  z.h[0] = (__bf16)a; z.h[1] = (__bf16)b;
  return z.u;
}

// A-tile (16x32 bf16) from LDS latent tile, rows stride 260 uints (520 bf16).
// ISA 16-bit A layout: lane<16 -> K {kb..kb+7, kb+16..kb+23}; lane>=16 -> +8.
static __device__ __forceinline__ v16bf load_a16(const unsigned* L0, int kb, int lane) {
  const int m  = lane & 15;
  const int kh = (lane & 16) ? 8 : 0;
  const unsigned* row = L0 + m * 260;
  union { uint4 q[2]; v16bf v; } u;
  u.q[0] = *(const uint4*)(row + ((kb + kh) >> 1));
  u.q[1] = *(const uint4*)(row + ((kb + kh + 16) >> 1));
  return u.v;
}

// B-tile (32x16 bf16) from transposed bf16 weights Wt[dk][do]:
// lane holds column N=nb+lane%16; K {kb..kb+7, kb+16..kb+23}, lane>=16 -> +8.
static __device__ __forceinline__ v16bf load_b16(const __bf16* Wt, int kb, int nb, int lane) {
  const int kh = (lane & 16) ? 8 : 0;
  const __bf16* col = Wt + nb + (lane & 15);
  v16bf v;
  #pragma unroll
  for (int s = 0; s < 8; ++s) {
    v[s]     = col[(size_t)(kb + kh + s) << 9];
    v[s + 8] = col[(size_t)(kb + kh + 16 + s) << 9];
  }
  return v;
}

static __device__ __forceinline__ v8f wmma_bf16f32(v16bf a, v16bf b, v8f c) {
  return __builtin_amdgcn_wmma_f32_16x16x32_bf16(false, a, false, b, (short)0, c, false, false);
}

#if defined(__HIP_DEVICE_COMPILE__) && __has_builtin(__builtin_amdgcn_wmma_f32_16x16x4_f32)
#define HAVE_WMMA_F32K4 1
static __device__ __forceinline__ v8f wmma_f32k4(v2f a, v2f b, v8f c) {
  return __builtin_amdgcn_wmma_f32_16x16x4_f32(false, a, false, b, (short)0, c, false, false);
}
#endif

// ---------------------------------------------------------------------------
// K0: weight transpose + f32 -> bf16 convert. 3 * 512 * 512 elements.
// ---------------------------------------------------------------------------
__global__ void __launch_bounds__(256) k_prep_w(
    const float* __restrict__ WQ, const float* __restrict__ WK, const float* __restrict__ WV,
    unsigned short* __restrict__ WQt, unsigned short* __restrict__ WKt, unsigned short* __restrict__ WVt)
{
  const int id   = blockIdx.x * 256 + threadIdx.x;   // 0 .. 3*2^18-1
  const int wsel = id >> 18;
  const int rem  = id & 262143;                      // = dof*512 + dk (dk fast -> coalesced read)
  const int dk   = rem & 511;
  const int dof  = rem >> 9;
  const float* S = (wsel == 0) ? WQ : ((wsel == 1) ? WK : WV);
  unsigned short* D = (wsel == 0) ? WQt : ((wsel == 1) ? WKt : WVt);
  union { __bf16 h; unsigned short u; } z;
  z.h = (__bf16)S[rem];
  D[dk * 512 + dof] = z.u;
}

// ---------------------------------------------------------------------------
// K1: per block: VC v's. For each v: Q,K[16 x 512] via bf16 WMMA (f32 acc,
// bias folded into C init), Gram G[i][j] += Q.K^T via f32 WMMA on the f32
// tiles, qsum[i] += Q^2. Deterministic per-block partial output (272 floats).
// ---------------------------------------------------------------------------
__global__ void __launch_bounds__(256) k_qk_gram(
    const float* __restrict__ latent,
    const unsigned short* __restrict__ WQt_u, const unsigned short* __restrict__ WKt_u,
    const float* __restrict__ bq, const float* __restrict__ bk,
    float* __restrict__ partials)
{
  __shared__ unsigned Llds[NV][16][260];   // bf16 latent tiles (padded rows)
  __shared__ float Qs[8][16][17];          // wave-private f32 Q tile scratch
  __shared__ float Ks[8][16][17];          // wave-private f32 K tile scratch
  __shared__ float Gd[8][256];             // cross-wave Gram reduce
  __shared__ float qsd[8][16];             // cross-wave qsum reduce

  const __bf16* WQt = (const __bf16*)WQt_u;
  const __bf16* WKt = (const __bf16*)WKt_u;

  const int t    = threadIdx.x;
  const int w    = t >> 5;
  const int lane = t & 31;
  const int ncol = lane & 15;
  const int mb   = (lane & 16) ? 8 : 0;
  const int v0   = blockIdx.x * VC;
  const int li   = t >> 4, lseg = t & 15;

  v8f G = {0.f, 0.f, 0.f, 0.f, 0.f, 0.f, 0.f, 0.f};
  float qsacc = 0.f;

  for (int g = 0; g < NGRP; ++g) {
    // ---- stage NV latent rows [16 x 512] f32 -> bf16 LDS -------------------
    #pragma unroll
    for (int vp = 0; vp < NV; ++vp) {
      const int v = v0 + g * NV + vp;
      const float4* src = (const float4*)(latent + ((size_t)li * VN + v) * DD + lseg * 32);
      #pragma unroll
      for (int it = 0; it < 4; ++it) {
        float4 x = src[it * 2];
        float4 y = src[it * 2 + 1];
        uint4 u;
        u.x = pkbf(x.x, x.y); u.y = pkbf(x.z, x.w);
        u.z = pkbf(y.x, y.y); u.w = pkbf(y.z, y.w);
        *(uint4*)&Llds[vp][li][lseg * 16 + it * 4] = u;
      }
    }
    __syncthreads();

    #pragma unroll
    for (int nt = 0; nt < 4; ++nt) {                 // wave w owns douts [nb, nb+16)
      const int nb  = (w * 4 + nt) * 16;
      const float bqv = bq[nb + ncol];
      const float bkv = bk[nb + ncol];
      v8f aq[NV], ak[NV];
      #pragma unroll
      for (int vp = 0; vp < NV; ++vp) {
        #pragma unroll
        for (int r = 0; r < 8; ++r) { aq[vp][r] = bqv; ak[vp][r] = bkv; }
      }
      #pragma unroll 4
      for (int ks = 0; ks < 16; ++ks) {              // K-dim: 512 = 16 x 32
        const int kb = ks * 32;
        v16bf Bq = load_b16(WQt, kb, nb, lane);
        v16bf Bk = load_b16(WKt, kb, nb, lane);
        #pragma unroll
        for (int vp = 0; vp < NV; ++vp) {
          v16bf A = load_a16(&Llds[vp][0][0], kb, lane);
          aq[vp] = wmma_bf16f32(A, Bq, aq[vp]);
          ak[vp] = wmma_bf16f32(A, Bk, ak[vp]);
        }
      }
      // ---- Gram + qsum from the f32 tiles (wave-private, in-order LDS) ----
      #pragma unroll
      for (int vp = 0; vp < NV; ++vp) {
        #pragma unroll
        for (int r = 0; r < 8; ++r) {
          Qs[w][mb + r][ncol] = aq[vp][r];
          Ks[w][mb + r][ncol] = ak[vp][r];
        }
        __builtin_amdgcn_fence(__ATOMIC_ACQ_REL, "wavefront");
#ifdef HAVE_WMMA_F32K4
        {
          const int hi2 = (lane & 16) ? 2 : 0;       // f32 A/B 16x4 K split
          #pragma unroll
          for (int kk = 0; kk < 4; ++kk) {
            const int ka = kk * 4 + hi2;
            v2f a, b;
            a.x = Qs[w][ncol][ka]; a.y = Qs[w][ncol][ka + 1];   // A[m=i][k=n]
            b.x = Ks[w][ncol][ka]; b.y = Ks[w][ncol][ka + 1];   // B[k=n][j]
            G = wmma_f32k4(a, b, G);
          }
        }
#else
        {
          #pragma unroll
          for (int r = 0; r < 8; ++r) {
            float s = G[r];
            #pragma unroll
            for (int c = 0; c < 16; ++c)
              s = fmaf(Qs[w][mb + r][c], Ks[w][ncol][c], s);
            G[r] = s;
          }
        }
#endif
        if (lane < 16) {
          float s = qsacc;
          #pragma unroll
          for (int c = 0; c < 16; ++c) { const float q = Qs[w][lane][c]; s = fmaf(q, q, s); }
          qsacc = s;
        }
        __builtin_amdgcn_fence(__ATOMIC_ACQ_REL, "wavefront");
      }
    }
    __syncthreads();   // all waves done reading Llds before next refill
  }

  // ---- deterministic cross-wave + per-block partial output ----------------
  #pragma unroll
  for (int r = 0; r < 8; ++r) Gd[w][(mb + r) * 16 + ncol] = G[r];
  if (lane < 16) qsd[w][lane] = qsacc;
  __syncthreads();

  float s = 0.f;
  #pragma unroll
  for (int ww = 0; ww < 8; ++ww) s += Gd[ww][t];
  partials[(size_t)blockIdx.x * 272 + t] = s;
  if (t < 16) {
    float s2 = 0.f;
    #pragma unroll
    for (int ww = 0; ww < 8; ++ww) s2 += qsd[ww][t];
    partials[(size_t)blockIdx.x * 272 + 256 + t] = s2;
  }
}

// ---------------------------------------------------------------------------
// K1b: one block. Reduce partials, P0 = sigmoid(G/||Q_i||),
// proj = 0.5*(softmax axis0 + softmax axis1), then 10x projection_p.
// ---------------------------------------------------------------------------
__global__ void __launch_bounds__(256) k_proj(const float* __restrict__ part,
                                              float* __restrict__ Pout)
{
  __shared__ float Gs[256], E0[256], E1[256];
  __shared__ float qs[16], cm[16], rm[16], cs[16], rs[16];
  const int t = threadIdx.x;
  const int i = t >> 4, j = t & 15;

  float g = 0.f;
  for (int b = 0; b < NB1; ++b) g += part[(size_t)b * 272 + t];
  Gs[t] = g;
  if (t < 16) {
    float q = 0.f;
    for (int b = 0; b < NB1; ++b) q += part[(size_t)b * 272 + 256 + t];
    qs[t] = sqrtf(q);
  }
  __syncthreads();

  Gs[t] = 1.f / (1.f + expf(-(Gs[t] / qs[i])));      // sigmoid(G/||Q_i||)
  __syncthreads();

  if (t < 16) {
    float m0 = -1e30f, m1 = -1e30f;
    for (int k = 0; k < 16; ++k) { m0 = fmaxf(m0, Gs[k * 16 + t]); m1 = fmaxf(m1, Gs[t * 16 + k]); }
    cm[t] = m0; rm[t] = m1;
  }
  __syncthreads();
  E0[t] = expf(Gs[t] - cm[j]);                       // softmax over i (axis 0)
  E1[t] = expf(Gs[t] - rm[i]);                       // softmax over j (axis 1)
  __syncthreads();
  if (t < 16) {
    float s0 = 0.f, s1 = 0.f;
    for (int k = 0; k < 16; ++k) { s0 += E0[k * 16 + t]; s1 += E1[t * 16 + k]; }
    cs[t] = s0; rs[t] = s1;
  }
  __syncthreads();
  Gs[t] = 0.5f * (E0[t] / cs[j] + E1[t] / rs[i]);
  __syncthreads();

  for (int it = 0; it < 10; ++it) {
    Gs[t] = fmaxf(Gs[t], 0.f);                       // relu
    __syncthreads();
    if (t < 16) { float s = 0.f; for (int k = 0; k < 16; ++k) s += Gs[t * 16 + k]; rs[t] = s; }
    __syncthreads();
    Gs[t] = Gs[t] - (rs[i] - 1.f) * (1.f / 16.f);    // row-mean correction
    __syncthreads();
    if (t < 16) { float s = 0.f; for (int k = 0; k < 16; ++k) s += Gs[k * 16 + t]; cs[t] = s; }
    __syncthreads();
    Gs[t] = Gs[t] - (cs[j] - 1.f) * (1.f / 16.f);    // col-mean correction
    __syncthreads();
  }
  Pout[t] = Gs[t];
}

// ---------------------------------------------------------------------------
// K2: recompute V via bf16 WMMA, mix out[j] = sum_i P[i][j] * V[i], store.
// Wave-autonomous mixing through wave-private LDS scratch.
// ---------------------------------------------------------------------------
__global__ void __launch_bounds__(256) k_v_out(
    const float* __restrict__ latent,
    const unsigned short* __restrict__ WVt_u, const float* __restrict__ bv,
    const float* __restrict__ Pg, float* __restrict__ out)
{
  __shared__ unsigned Llds[NV][16][260];
  __shared__ float Vsw[8][16][17];
  __shared__ float Pl[256];

  const __bf16* WVt = (const __bf16*)WVt_u;
  const int t    = threadIdx.x;
  const int w    = t >> 5;
  const int lane = t & 31;
  const int ncol = lane & 15;
  const int mb   = (lane & 16) ? 8 : 0;
  const int half = lane >> 4;
  const int v0   = blockIdx.x * VC;
  const int li   = t >> 4, lseg = t & 15;

  Pl[t] = Pg[t];
  __syncthreads();
  float pc[16];                                       // P column j = ncol
  #pragma unroll
  for (int i = 0; i < 16; ++i) pc[i] = Pl[i * 16 + ncol];

  for (int g = 0; g < NGRP; ++g) {
    #pragma unroll
    for (int vp = 0; vp < NV; ++vp) {
      const int v = v0 + g * NV + vp;
      const float4* src = (const float4*)(latent + ((size_t)li * VN + v) * DD + lseg * 32);
      #pragma unroll
      for (int it = 0; it < 4; ++it) {
        float4 x = src[it * 2];
        float4 y = src[it * 2 + 1];
        uint4 u;
        u.x = pkbf(x.x, x.y); u.y = pkbf(x.z, x.w);
        u.z = pkbf(y.x, y.y); u.w = pkbf(y.z, y.w);
        *(uint4*)&Llds[vp][li][lseg * 16 + it * 4] = u;
      }
    }
    __syncthreads();

    #pragma unroll
    for (int nt = 0; nt < 4; ++nt) {
      const int nb  = (w * 4 + nt) * 16;
      const float bvv = bv[nb + ncol];
      v8f av[NV];
      #pragma unroll
      for (int vp = 0; vp < NV; ++vp) {
        #pragma unroll
        for (int r = 0; r < 8; ++r) av[vp][r] = bvv;
      }
      #pragma unroll 4
      for (int ks = 0; ks < 16; ++ks) {
        const int kb = ks * 32;
        v16bf Bv = load_b16(WVt, kb, nb, lane);
        #pragma unroll
        for (int vp = 0; vp < NV; ++vp) {
          v16bf A = load_a16(&Llds[vp][0][0], kb, lane);
          av[vp] = wmma_bf16f32(A, Bv, av[vp]);
        }
      }
      #pragma unroll
      for (int vp = 0; vp < NV; ++vp) {
        const int v = v0 + g * NV + vp;
        #pragma unroll
        for (int r = 0; r < 8; ++r) Vsw[w][mb + r][ncol] = av[vp][r];
        __builtin_amdgcn_fence(__ATOMIC_ACQ_REL, "wavefront");
        float acc[8];
        #pragma unroll
        for (int oo = 0; oo < 8; ++oo) acc[oo] = 0.f;
        #pragma unroll
        for (int i = 0; i < 16; ++i) {
          const float pv = pc[i];
          #pragma unroll
          for (int oo = 0; oo < 8; ++oo)
            acc[oo] = fmaf(pv, Vsw[w][i][half * 8 + oo], acc[oo]);
        }
        float* op = out + ((size_t)ncol * VN + v) * DD + nb + half * 8;
        float4 o0 = {acc[0], acc[1], acc[2], acc[3]};
        float4 o1 = {acc[4], acc[5], acc[6], acc[7]};
        *(float4*)op       = o0;
        *(float4*)(op + 4) = o1;
        __builtin_amdgcn_fence(__ATOMIC_ACQ_REL, "wavefront");
      }
    }
    __syncthreads();
  }
}

// ---------------------------------------------------------------------------
extern "C" void kernel_launch(void* const* d_in, const int* in_sizes, int n_in,
                              void* d_out, int out_size, void* d_ws, size_t ws_size,
                              hipStream_t stream)
{
  (void)in_sizes; (void)n_in; (void)out_size; (void)ws_size;
  const float* latent = (const float*)d_in[0];
  const float* WQ = (const float*)d_in[1];
  const float* bq = (const float*)d_in[2];
  const float* WK = (const float*)d_in[3];
  const float* bk = (const float*)d_in[4];
  const float* WV = (const float*)d_in[5];
  const float* bv = (const float*)d_in[6];
  float* out = (float*)d_out;

  char* ws = (char*)d_ws;
  const size_t WBYTES = (size_t)512 * 512 * 2;       // one bf16 weight matrix
  unsigned short* WQt = (unsigned short*)(ws);
  unsigned short* WKt = (unsigned short*)(ws + WBYTES);
  unsigned short* WVt = (unsigned short*)(ws + 2 * WBYTES);
  float* partials     = (float*)(ws + 3 * WBYTES);                    // NB1 * 272 f32
  float* P            = (float*)(ws + 3 * WBYTES + (size_t)NB1 * 272 * 4);  // 256 f32

  k_prep_w<<<dim3(3 * 262144 / 256), dim3(256), 0, stream>>>(WQ, WK, WV, WQt, WKt, WVt);
  k_qk_gram<<<dim3(NB1), dim3(256), 0, stream>>>(latent, WQt, WKt, bq, bk, partials);
  k_proj<<<dim3(1), dim3(256), 0, stream>>>(partials, P);
  k_v_out<<<dim3(NB1), dim3(256), 0, stream>>>(latent, WVt, bv, P, out);
}
